// DGVAE_8942121910580
// MI455X (gfx1250) — matrix operand order repaired and checked by reading, not compiled, not run on hardware
//
#include <hip/hip_runtime.h>
#include <hip/hip_bf16.h>

typedef __attribute__((ext_vector_type(16))) _Float16 v16h;
typedef __attribute__((ext_vector_type(8)))  float    v8f;

union Frag16 { v16h v; _Float16 h[16]; unsigned u[8]; };
union Frag8f { v8f  v; float f[8]; };
union HalfPair { unsigned u; _Float16 h[2]; };

// ---------------------------------------------------------------------------
// Deterministic "permutation" generation (stand-in for jax.random.permutation)
// p[0..9]   = hop-1 column sample (10)
// p[16..40] = hop-2 column sample (25)
// ---------------------------------------------------------------------------
__global__ void build_perm_kernel(int* __restrict__ p) {
  if (threadIdx.x != 0 || blockIdx.x != 0) return;
  int a[128];
  unsigned s = 42u * 747796405u + 2891336453u;
  for (int i = 0; i < 128; ++i) a[i] = i;
  for (int i = 127; i > 0; --i) {
    s = s * 1664525u + 1013904223u;
    int j = (int)((s >> 8) % (unsigned)(i + 1));
    int t = a[i]; a[i] = a[j]; a[j] = t;
  }
  for (int i = 0; i < 10; ++i) p[i] = a[i];
  for (int i = 0; i < 128; ++i) a[i] = i;
  s ^= 0x9E3779B9u;
  for (int i = 127; i > 0; --i) {
    s = s * 1664525u + 1013904223u;
    int j = (int)((s >> 8) % (unsigned)(i + 1));
    int t = a[i]; a[i] = a[j]; a[j] = t;
  }
  for (int i = 0; i < 25; ++i) p[16 + i] = a[i];
}

// out[i*S+j] = adj[src[i]*128 + perm[j]]   (S is compile-time: no div emulation)
template <int S>
__global__ __launch_bounds__(256) void sample_kernel(
    const int* __restrict__ adj, const int* __restrict__ src,
    const int* __restrict__ perm, int* __restrict__ out, int nsrc) {
  int t = blockIdx.x * blockDim.x + threadIdx.x;
  if (t >= nsrc * S) return;
  int i = t / S, j = t - i * S;
  out[t] = adj[(size_t)src[i] * 128 + perm[j]];
}

// ---------------------------------------------------------------------------
// Degree-normalized mean aggregation over f32 features -> f16 rows
// one wave per row, lane owns 4 columns (float4, fully coalesced, L2-resident)
// ---------------------------------------------------------------------------
__global__ __launch_bounds__(256) void agg_feat_kernel(
    const float* __restrict__ features, const int* __restrict__ selfIdx,
    const int* __restrict__ nbIdx, const float* __restrict__ degrees,
    _Float16* __restrict__ out, int S, int nrows) {
  int row  = blockIdx.x * 8 + (threadIdx.x >> 5);
  int lane = threadIdx.x & 31;
  if (row >= nrows) return;
  float4 s = ((const float4*)(features + (size_t)selfIdx[row] * 128))[lane];
  float ax = 0.f, ay = 0.f, az = 0.f, aw = 0.f;
  const int* nb = nbIdx + (size_t)row * S;
  for (int j = 0; j < S; ++j) {
    int v = nb[j];
    if (j + 1 < S) __builtin_prefetch(features + (size_t)nb[j + 1] * 128, 0, 1);
    float inv = 1.0f / degrees[v];
    float4 f = ((const float4*)(features + (size_t)v * 128))[lane];
    ax += f.x * inv; ay += f.y * inv; az += f.z * inv; aw += f.w * inv;
  }
  float invS = 1.0f / (float)S;
  size_t o = (size_t)row * 128 + (size_t)lane * 4;
  out[o + 0] = (_Float16)(s.x + ax * invS);
  out[o + 1] = (_Float16)(s.y + ay * invS);
  out[o + 2] = (_Float16)(s.z + az * invS);
  out[o + 3] = (_Float16)(s.w + aw * invS);
}

// hagg = h0 + mean_j(h1[row*S+j]/deg(nb1[row*S+j]))   (f16 in / f16 out)
__global__ __launch_bounds__(256) void agg_h_kernel(
    const _Float16* __restrict__ h1, const _Float16* __restrict__ h0,
    const int* __restrict__ nb1, const float* __restrict__ degrees,
    _Float16* __restrict__ out, int S, int nrows) {
  int row  = blockIdx.x * 8 + (threadIdx.x >> 5);
  int lane = threadIdx.x & 31;
  if (row >= nrows) return;
  int col = lane * 4;
  float a0 = 0.f, a1 = 0.f, a2 = 0.f, a3 = 0.f;
  for (int j = 0; j < S; ++j) {
    int r = row * S + j;
    float inv = 1.0f / degrees[nb1[r]];
    const _Float16* p = h1 + (size_t)r * 128 + col;
    a0 += (float)p[0] * inv; a1 += (float)p[1] * inv;
    a2 += (float)p[2] * inv; a3 += (float)p[3] * inv;
  }
  float invS = 1.0f / (float)S;
  const _Float16* sp = h0 + (size_t)row * 128 + col;
  _Float16* op = out + (size_t)row * 128 + col;
  op[0] = (_Float16)((float)sp[0] + a0 * invS);
  op[1] = (_Float16)((float)sp[1] + a1 * invS);
  op[2] = (_Float16)((float)sp[2] + a2 * invS);
  op[3] = (_Float16)((float)sp[3] + a3 * invS);
}

// ---------------------------------------------------------------------------
// WMMA GEMM, fully compile-time tiled: out[M,N] = act(A[M,K]*B[K,N] + bias)
// 8 waves / block, 64-row stripe. K consumed in 32-wide v_wmma steps.
// All tile/k loops unrolled -> accumulators live in VGPRs (no scratch).
// B is staged transposed in LDS with +2-half pad so each B fragment is
// 8 x ds_load_b32 and strided accesses spread across the 64 banks.
// Fragment layouts per CDNA5 ISA 7.12.2 (wave32).
// ---------------------------------------------------------------------------
template <int KT, int NT, int ACT, int OUTF32>
__global__ __launch_bounds__(256) void gemm_wmma_kernel(
    const _Float16* __restrict__ A, const _Float16* __restrict__ B,
    const float* __restrict__ bias, void* __restrict__ out, int M) {
  constexpr int KC0 = 16384 / NT;
  constexpr int KC1 = (KC0 > 128) ? 128 : KC0;
  constexpr int KC  = (KC1 > KT) ? KT : KC1;      // K-chunk staged in LDS
  constexpr int NCT = NT / 16;                    // 16-wide col tiles
  constexpr int TOTAL = 4 * NCT;                  // (64/16) row tiles * NCT
  constexpr int NPER  = TOTAL / 8;                // tiles per wave (<= 8)
  constexpr int KSTEPS = KC / 32;
  constexpr int SBS = KC + 2;                     // padded LDS stride (halves)

  __shared__ _Float16 sA[64 * KC];
  __shared__ _Float16 sBt[NT * SBS];

  const int tid  = threadIdx.x;
  const int wave = tid >> 5;
  const int lane = tid & 31;
  const int half = lane >> 4;
  const int lm   = lane & 15;
  const int m0   = blockIdx.x * 64;

  Frag8f acc[NPER];
#pragma unroll
  for (int l = 0; l < NPER; ++l)
#pragma unroll
    for (int r = 0; r < 8; ++r) acc[l].f[r] = 0.f;

  for (int kc = 0; kc < KT; kc += KC) {
    { // B chunk -> transposed LDS: sBt[n*SBS + k] = B[kc+k][n]
      const unsigned* gB = (const unsigned*)(B + (size_t)kc * NT);
      constexpr int CNT = KC * NT / 2;
      for (int i = tid; i < CNT; i += 256) {
        int k = i / (NT / 2);
        int n = (i - k * (NT / 2)) * 2;
        HalfPair t; t.u = gB[i];
        sBt[(size_t)n * SBS + k]       = t.h[0];
        sBt[(size_t)(n + 1) * SBS + k] = t.h[1];
      }
    }
    { // A chunk: 64 rows x KC cols (row-major in LDS)
      unsigned* dA = (unsigned*)sA;
      constexpr int CPR = KC / 2;
      constexpr int CNT = 64 * CPR;
      for (int i = tid; i < CNT; i += 256) {
        int r = i / CPR, c = i - r * CPR;
        int gm = m0 + r;
        dA[i] = (gm < M) ? ((const unsigned*)(A + (size_t)gm * KT + kc))[c] : 0u;
      }
    }
    __syncthreads();

#pragma unroll
    for (int l = 0; l < NPER; ++l) {
      const int t  = wave + (l << 3);
      const int rt = t / NCT, ct = t - rt * NCT;
      const _Float16* pa = sA  + (size_t)(rt * 16 + lm) * KC;
      const _Float16* pb = sBt + (size_t)(ct * 16 + lm) * SBS;
#pragma unroll
      for (int ks = 0; ks < KSTEPS; ++ks) {
        Frag16 a, b;
        const int kbA = (ks << 5) + (half << 3);
#pragma unroll
        for (int i = 0; i < 8; ++i) {
          int k0 = kbA + ((i >> 2) << 4) + ((i & 3) << 1);
          a.u[i] = *(const unsigned*)(pa + k0);
        }
        const int kbB = (ks << 5) + (half << 3);
#pragma unroll
        for (int i = 0; i < 8; ++i)
          b.u[i] = *(const unsigned*)(pb + kbB + ((i >> 2) << 4) + ((i & 3) << 1));
        acc[l].v = __builtin_amdgcn_wmma_f32_16x16x32_f16(
            false, a.v, false, b.v, (short)0, acc[l].v, false, false);
      }
    }
    __syncthreads();
  }

  // epilogue: bias + activation + store (f16 or f32), all compile-time paths
#pragma unroll
  for (int l = 0; l < NPER; ++l) {
    const int t  = wave + (l << 3);
    const int rt = t / NCT, ct = t - rt * NCT;
    const int n  = (ct << 4) + lm;
    const int mb = m0 + rt * 16 + (half << 3);
    const float bv = bias ? bias[n] : 0.f;
#pragma unroll
    for (int r = 0; r < 8; ++r) {
      int gm = mb + r;
      if (gm >= M) continue;
      float v = acc[l].f[r] + bv;
      if (ACT == 1)      v = tanhf(v);
      else if (ACT == 2) v = fmaxf(v, 0.f);
      if (OUTF32) ((float*)out)[(size_t)gm * NT + n] = v;
      else ((_Float16*)out)[(size_t)gm * NT + n] = (_Float16)v;
    }
  }
}

// ---------------------------------------------------------------------------
// weight conversion / padding
// ---------------------------------------------------------------------------
__global__ __launch_bounds__(256) void f32_to_f16_kernel(
    const float* __restrict__ s, _Float16* __restrict__ d, int n) {
  int i = blockIdx.x * blockDim.x + threadIdx.x;
  if (i < n) d[i] = (_Float16)s[i];
}

__global__ __launch_bounds__(256) void pad_w2_kernel(
    const float* __restrict__ W2, _Float16* __restrict__ d) {
  int t = blockIdx.x * blockDim.x + threadIdx.x;       // 256*64
  if (t >= 256 * 64) return;
  int r = t >> 6, c = t & 63;
  d[t] = (_Float16)((c < 50) ? W2[r * 50 + c] : 0.f);
}

__global__ void pad_b2_kernel(const float* __restrict__ b2, float* __restrict__ d) {
  int t = threadIdx.x;
  if (t < 64) d[t] = (t < 50) ? b2[t] : 0.f;
}

// ---------------------------------------------------------------------------
// reparameterized sample: z = mean + eps*exp(logstd), eps ~ N(0,1) (hash BM)
// ---------------------------------------------------------------------------
__device__ __forceinline__ unsigned hash_u32(unsigned x) {
  x ^= x >> 16; x *= 0x7feb352dU; x ^= x >> 15; x *= 0x846ca68bU; x ^= x >> 16;
  return x;
}

__global__ __launch_bounds__(256) void reparam_kernel(
    const float* __restrict__ zm, const float* __restrict__ zs,
    _Float16* __restrict__ zh, int n) {
  int i = blockIdx.x * blockDim.x + threadIdx.x;
  if (i >= n) return;
  unsigned h1 = hash_u32(0x9E3779B9u ^ (unsigned)i);
  unsigned h2 = hash_u32(0x85EBCA6Bu + (unsigned)i);
  float u1 = ((float)(h1 >> 8) + 1.0f) * (1.0f / 16777217.0f);
  float u2 = (float)(h2 >> 8) * (1.0f / 16777216.0f);
  float r  = __fsqrt_rn(-2.0f * __logf(u1));
  float eps = r * __cosf(6.28318530718f * u2);
  zh[i] = (_Float16)(zm[i] + eps * __expf(zs[i]));
}

// ---------------------------------------------------------------------------
// softmax over first 50 of 64 padded logits; one wave per row
// ---------------------------------------------------------------------------
__global__ __launch_bounds__(256) void softmax_kernel(
    const float* __restrict__ logits, float* __restrict__ out, int rows) {
  int row  = blockIdx.x * 8 + (threadIdx.x >> 5);
  int lane = threadIdx.x & 31;
  if (row >= rows) return;
  const float* lr = logits + (size_t)row * 64;
  float v0 = (lane < 50)      ? lr[lane]      : -1e30f;
  float v1 = (lane + 32 < 50) ? lr[lane + 32] : -1e30f;
  float m = fmaxf(v0, v1);
  for (int off = 16; off; off >>= 1) m = fmaxf(m, __shfl_xor(m, off, 32));
  float e0 = (lane < 50)      ? __expf(v0 - m) : 0.f;
  float e1 = (lane + 32 < 50) ? __expf(v1 - m) : 0.f;
  float sum = e0 + e1;
  for (int off = 16; off; off >>= 1) sum += __shfl_xor(sum, off, 32);
  float inv = 1.0f / sum;
  if (lane < 50)      out[(size_t)row * 50 + lane]      = e0 * inv;
  if (lane + 32 < 50) out[(size_t)row * 50 + lane + 32] = e1 * inv;
}

// ---------------------------------------------------------------------------
// launch
// ---------------------------------------------------------------------------
static inline size_t align256(size_t x) { return (x + 255) & ~(size_t)255; }

extern "C" void kernel_launch(void* const* d_in, const int* in_sizes, int n_in,
                              void* d_out, int out_size, void* d_ws, size_t ws_size,
                              hipStream_t stream) {
  (void)in_sizes; (void)n_in; (void)out_size; (void)ws_size;
  const int*   nodes    = (const int*)d_in[0];
  const int*   adj      = (const int*)d_in[1];
  const float* degrees  = (const float*)d_in[2];
  const float* features = (const float*)d_in[3];
  const float* W0       = (const float*)d_in[4];
  const float* Wm       = (const float*)d_in[5];
  const float* Ws       = (const float*)d_in[6];
  const float* W1       = (const float*)d_in[7];
  const float* b1       = (const float*)d_in[8];
  const float* W2       = (const float*)d_in[9];
  const float* b2       = (const float*)d_in[10];

  const int Bn = 4096;
  const int NB1 = Bn * 10;          // 40960 hop-1 samples
  const int NB2 = NB1 * 25;         // 1,024,000 hop-2 samples

  char* w = (char*)d_ws; size_t off = 0;
  auto carve = [&](size_t bytes) { void* p = w + off; off = align256(off + bytes); return p; };
  int*      perm = (int*)carve(64 * sizeof(int));
  int*      nb1  = (int*)carve((size_t)NB1 * sizeof(int));
  int*      nb2  = (int*)carve((size_t)NB2 * sizeof(int));
  _Float16* W0h  = (_Float16*)carve(128 * 128 * 2);
  _Float16* Wmh  = (_Float16*)carve(128 * 64 * 2);
  _Float16* Wsh  = (_Float16*)carve(128 * 64 * 2);
  _Float16* W1h  = (_Float16*)carve(64 * 256 * 2);
  _Float16* W2h  = (_Float16*)carve(256 * 64 * 2);
  float*    b2p  = (float*)carve(64 * sizeof(float));
  _Float16* agg1 = (_Float16*)carve((size_t)NB1 * 128 * 2);
  _Float16* agg0 = (_Float16*)carve((size_t)Bn * 128 * 2);
  _Float16* hagg = (_Float16*)carve((size_t)Bn * 128 * 2);
  float*    zm   = (float*)carve((size_t)Bn * 64 * sizeof(float));
  float*    zs   = (float*)carve((size_t)Bn * 64 * sizeof(float));
  _Float16* zh   = (_Float16*)carve((size_t)Bn * 64 * 2);
  _Float16* d1h  = (_Float16*)carve((size_t)Bn * 256 * 2);
  float*    logits = (float*)carve((size_t)Bn * 64 * sizeof(float));
  _Float16* h1 = agg1;   // in-place GEMM (single K-chunk: safe)
  _Float16* h0 = agg0;

  // sampling
  build_perm_kernel<<<1, 32, 0, stream>>>(perm);
  sample_kernel<10><<<(NB1 + 255) / 256, 256, 0, stream>>>(adj, nodes, perm, nb1, Bn);
  sample_kernel<25><<<(NB2 + 255) / 256, 256, 0, stream>>>(adj, nb1, perm + 16, nb2, NB1);

  // weight precision conversion
  f32_to_f16_kernel<<<64, 256, 0, stream>>>(W0, W0h, 128 * 128);
  f32_to_f16_kernel<<<32, 256, 0, stream>>>(Wm, Wmh, 128 * 64);
  f32_to_f16_kernel<<<32, 256, 0, stream>>>(Ws, Wsh, 128 * 64);
  f32_to_f16_kernel<<<64, 256, 0, stream>>>(W1, W1h, 64 * 256);
  pad_w2_kernel<<<64, 256, 0, stream>>>(W2, W2h);
  pad_b2_kernel<<<1, 64, 0, stream>>>(b2, b2p);

  // layer 0: aggregate + shared-W0 conv with tanh (WMMA)
  agg_feat_kernel<<<NB1 / 8, 256, 0, stream>>>(features, nb1, nb2, degrees, agg1, 25, NB1);
  agg_feat_kernel<<<Bn / 8, 256, 0, stream>>>(features, nodes, nb1, degrees, agg0, 10, Bn);
  gemm_wmma_kernel<128, 128, 1, 0><<<NB1 / 64, 256, 0, stream>>>(agg1, W0h, nullptr, h1, NB1);
  gemm_wmma_kernel<128, 128, 1, 0><<<Bn / 64, 256, 0, stream>>>(agg0, W0h, nullptr, h0, Bn);

  // layer 1: mean/std heads on hop-0 (WMMA), then reparameterize
  agg_h_kernel<<<Bn / 8, 256, 0, stream>>>(h1, h0, nb1, degrees, hagg, 10, Bn);
  gemm_wmma_kernel<128, 64, 0, 1><<<Bn / 64, 256, 0, stream>>>(hagg, Wmh, nullptr, zm, Bn);
  gemm_wmma_kernel<128, 64, 0, 1><<<Bn / 64, 256, 0, stream>>>(hagg, Wsh, nullptr, zs, Bn);
  reparam_kernel<<<(Bn * 64 + 255) / 256, 256, 0, stream>>>(zm, zs, zh, Bn * 64);

  // decoder MLP (WMMA) + softmax
  gemm_wmma_kernel<64, 256, 2, 0><<<Bn / 64, 256, 0, stream>>>(zh, W1h, b1, d1h, Bn);
  gemm_wmma_kernel<256, 64, 0, 1><<<Bn / 64, 256, 0, stream>>>(d1h, W2h, b2p, logits, Bn);
  softmax_kernel<<<Bn / 8, 256, 0, stream>>>(logits, (float*)d_out, Bn);
}
